// SwinTransformerBlock_71528385348223
// MI455X (gfx1250) — matrix operand order repaired and checked by reading, not compiled, not run on hardware
//
#include <hip/hip_runtime.h>
#include <math.h>

// ---------------------------------------------------------------------------
// Swin transformer block, fused, for gfx1250 (MI455X).
// All GEMMs via v_wmma_f32_16x16x32_bf16 (fp32 accumulate).
// Weights pre-converted fp32->bf16 AND pre-swizzled into WMMA B-fragment
// order so every lane loads its 16 fragment elements as one contiguous
// 32-byte run (global_load_b128 pairs, fully coalesced per wave).
// ---------------------------------------------------------------------------

typedef __bf16 bf16;
typedef __attribute__((ext_vector_type(16))) __bf16 v16bf;
typedef __attribute__((ext_vector_type(8)))  float  v8f;

#define B_    64
#define IS_   56
#define WS_   7
#define SS_   3
#define C_    192
#define NH_   6
#define N_    49
#define HD_   32
#define TOK_  (IS_*IS_)          // 3136
#define SCALE_ 0.17677669529663687f  // 1/sqrt(32)

// bf16 element counts of the four swizzled weight blobs in workspace
#define W_QKV_OFF  0
#define W_PROJ_OFF 110592
#define W_FC1_OFF  147456
#define W_FC2_OFF  294912
#define W_TOTAL    442368

union AF { v16bf v; bf16 e[16]; };
union CF { v8f  v; float e[8]; };

__device__ __forceinline__ v8f zero8() {
  v8f z = {0.f, 0.f, 0.f, 0.f, 0.f, 0.f, 0.f, 0.f};
  return z;
}

__device__ __forceinline__ v8f wmma_bf(v16bf a, v16bf b, v8f c) {
  // 8 args: (neg_a, A, neg_b, B, c_mod, C, reuse_a, reuse_b)
  return __builtin_amdgcn_wmma_f32_16x16x32_bf16(false, a, false, b, (short)0, c, false, false);
}

// A fragment: 16x32 bf16 tile, row-major source A[row][col], lda elements.
// Layout: lane L: m=L&15, half=L>>4; element j holds
//   k = half*16 + j
// -> per lane this is one contiguous 16-element run (matches B fragments).
__device__ __forceinline__ v16bf load_a_frag(const bf16* A, int lda, int r0, int k0, int valid_rows) {
  int lane = threadIdx.x & 31;
  int m = lane & 15, half = lane >> 4;
  AF f;
  int row = r0 + m;
  const bf16* base = A + (size_t)row * lda + k0;
  bool ok = row < valid_rows;
#pragma unroll
  for (int j = 0; j < 16; ++j) {
    int kl = half * 16 + j;
    f.e[j] = ok ? base[kl] : (bf16)0.0f;
  }
  return f.v;
}

// B fragment: 32x16 bf16 tile from row-major B[k][n], ldb elements (LDS source).
// Lane L: n=L&15; K values (L>>4)*16 + j, j=0..15 in vector order.
__device__ __forceinline__ v16bf load_b_frag(const bf16* Bp, int ldb, int k0, int n0) {
  int lane = threadIdx.x & 31;
  int n = lane & 15, half = lane >> 4;
  AF f;
  const bf16* base = Bp + (size_t)(k0 + half * 16) * ldb + n0 + n;
#pragma unroll
  for (int j = 0; j < 16; ++j) f.e[j] = base[(size_t)j * ldb];
  return f.v;
}

// B fragment from pre-swizzled global weights: tile (kt,nt) stored as 512
// consecutive bf16, lane-major (lane*16 + j). One 32B contiguous read/lane.
__device__ __forceinline__ v16bf load_b_frag_sw(const bf16* __restrict__ Wsw, int ntn, int kt, int nt) {
  const bf16* p = Wsw + (((size_t)kt * ntn + nt) << 9) + ((threadIdx.x & 31) << 4);
  AF f;
#pragma unroll
  for (int j = 0; j < 16; ++j) f.e[j] = p[j];
  return f.v;
}

// B fragment from a TRANSPOSED LDS source: logical B[k][n] = Kp[n*ld + k].
// Per-lane contiguous 16-element read.
__device__ __forceinline__ v16bf load_bt_frag(const bf16* Kp, int ld, int k0, int n0) {
  int lane = threadIdx.x & 31;
  int n = lane & 15, half = lane >> 4;
  AF f;
  const bf16* base = Kp + (size_t)(n0 + n) * ld + k0 + half * 16;
#pragma unroll
  for (int j = 0; j < 16; ++j) f.e[j] = base[j];
  return f.v;
}

__device__ __forceinline__ int region_(int p) { return p < 49 ? 0 : (p < 53 ? 1 : 2); }

// ---------------------------------------------------------------------------
// Kernel 0: convert fp32 weights -> bf16 with B-fragment swizzle.
// Destination index i decomposes as: tile = i/512, lane = (i%512)/16, j = i%16
//   kt = tile / (Nc/16), nt = tile % (Nc/16)
//   source element = W[kt*32 + (lane>>4)*16 + j][nt*16 + (lane&15)]
// ---------------------------------------------------------------------------
__device__ __forceinline__ void conv_swizzle(const float* __restrict__ w, bf16* __restrict__ o,
                                             int Nc, int i) {
  int tile = i >> 9, rem = i & 511;
  int lane = rem >> 4, j = rem & 15;
  int ntn = Nc >> 4;
  int kt = tile / ntn, nt = tile - kt * ntn;
  int k = kt * 32 + ((lane >> 4) << 4) + j;
  int n = (nt << 4) + (lane & 15);
  o[i] = (bf16)w[(size_t)k * Nc + n];
}

__global__ void swin_convert_w_kernel(const float* __restrict__ qkvw, const float* __restrict__ projw,
                                      const float* __restrict__ fc1w, const float* __restrict__ fc2w,
                                      bf16* __restrict__ o) {
  int i = blockIdx.x * blockDim.x + threadIdx.x;
  if (i < W_PROJ_OFF)      conv_swizzle(qkvw,  o + W_QKV_OFF,  576, i - W_QKV_OFF);
  else if (i < W_FC1_OFF)  conv_swizzle(projw, o + W_PROJ_OFF, 192, i - W_PROJ_OFF);
  else if (i < W_FC2_OFF)  conv_swizzle(fc1w,  o + W_FC1_OFF,  768, i - W_FC1_OFF);
  else if (i < W_TOTAL)    conv_swizzle(fc2w,  o + W_FC2_OFF,  192, i - W_FC2_OFF);
}

// ---------------------------------------------------------------------------
// Kernel 1: fused  LN1 -> shift+window gather -> QKV -> attention -> proj
//           -> inverse shift scatter + residual.   One block per window.
// Writes x1 = x + attn_out into d_out (fp32).
// ---------------------------------------------------------------------------
#define SM_Q   0
#define SM_K   (SM_Q   + 64*C_*2)
#define SM_V   (SM_K   + 64*C_*2)
#define SM_ALN (SM_V   + 64*C_*2)       // reused as obuf after QKV
#define SM_S   (SM_ALN + 64*C_*2)
#define SM_P   (SM_S   + 49*64*4)
#define SM_TOT (SM_P   + 49*64*2)       // 117120 bytes

__global__ __launch_bounds__(256, 1) void swin_attn_kernel(
    const float* __restrict__ x, const float* __restrict__ n1w, const float* __restrict__ n1b,
    const bf16* __restrict__ qkvw, const float* __restrict__ qkvb, const float* __restrict__ rpb,
    const bf16* __restrict__ projw, const float* __restrict__ projb, float* __restrict__ x1)
{
  __shared__ __align__(16) char smem[SM_TOT];
  bf16*  qbuf = (bf16*)(smem + SM_Q);    // [64][192], head h at cols h*32..
  bf16*  kbuf = (bf16*)(smem + SM_K);
  bf16*  vbuf = (bf16*)(smem + SM_V);
  bf16*  aln  = (bf16*)(smem + SM_ALN);  // LN'd input tile [64][192]
  bf16*  obuf = aln;                     // reused for concat-head output
  float* Sb   = (float*)(smem + SM_S);   // per-head scores [49][64]
  bf16*  Pb   = (bf16*)(smem + SM_P);    // per-head probs  [49][64]

  int w  = blockIdx.x & 63;
  int b  = blockIdx.x >> 6;
  int wr = w >> 3, wc = w & 7;
  const float* xb  = x  + (size_t)b * TOK_ * C_;
  float*       x1b = x1 + (size_t)b * TOK_ * C_;
  int wave = threadIdx.x >> 5, lane = threadIdx.x & 31;

  // ---- phase 1: gather rolled window rows, LayerNorm1, store bf16 ----
  for (int t = wave; t < N_; t += 8) {
    int tr = t / 7, tc = t % 7;
    int gr = (wr * 7 + tr + SS_) % IS_;   // rolled -> original coords
    int gc = (wc * 7 + tc + SS_) % IS_;
    const float* row = xb + (size_t)(gr * IS_ + gc) * C_;
    float vals[6]; float s = 0.f, s2 = 0.f;
#pragma unroll
    for (int i = 0; i < 6; ++i) { float v = row[lane + i * 32]; vals[i] = v; s += v; s2 += v * v; }
#pragma unroll
    for (int off = 16; off > 0; off >>= 1) { s += __shfl_xor(s, off, 32); s2 += __shfl_xor(s2, off, 32); }
    float mu   = s  * (1.f / C_);
    float var  = s2 * (1.f / C_) - mu * mu;
    float rstd = rsqrtf(var + 1e-5f);
#pragma unroll
    for (int i = 0; i < 6; ++i) {
      int c = lane + i * 32;
      aln[t * C_ + c] = (bf16)((vals[i] - mu) * rstd * n1w[c] + n1b[c]);
    }
  }
  for (int i = threadIdx.x; i < 15 * C_; i += 256) aln[N_ * C_ + i] = (bf16)0.0f; // pad rows
  __syncthreads();

  // ---- phase 2: QKV gemm  [64x192] @ [192x576] ----
  for (int nt = wave; nt < 36; nt += 8) {
    v16bf bfr[6];
#pragma unroll
    for (int kt = 0; kt < 6; ++kt) bfr[kt] = load_b_frag_sw(qkvw, 36, kt, nt);
    int n = lane & 15, hm = lane >> 4;
    int col = nt * 16 + n;
    int which = col / C_, cc = col % C_;
    float bias = qkvb[col];
    bf16* dst = (which == 0) ? qbuf : (which == 1) ? kbuf : vbuf;
    float sc  = (which == 0) ? SCALE_ : 1.f;
    for (int mt = 0; mt < 4; ++mt) {
      v8f acc = zero8();
#pragma unroll
      for (int kt = 0; kt < 6; ++kt) {
        v16bf afr = load_a_frag(aln, C_, mt * 16, kt * 32, 64);
        acc = wmma_bf(afr, bfr[kt], acc);
      }
      CF c; c.v = acc;
#pragma unroll
      for (int r = 0; r < 8; ++r) {
        int m = mt * 16 + r + hm * 8;
        dst[m * C_ + cc] = (bf16)((c.e[r] + bias) * sc);
      }
    }
  }
  __syncthreads();

  // ---- phase 3..5: per-head attention ----
  for (int h = 0; h < NH_; ++h) {
    // S = (q*scale) @ k^T + rel_pos_bias + shift_mask
    for (int job = wave; job < 16; job += 8) {
      int mt = job >> 2, nt2 = job & 3;
      v16bf afr = load_a_frag(qbuf + h * HD_, C_, mt * 16, 0, 64);
      v16bf bfr = load_bt_frag(kbuf + h * HD_, C_, 0, nt2 * 16);
      v8f acc = wmma_bf(afr, bfr, zero8());
      CF c; c.v = acc;
      int n = nt2 * 16 + (lane & 15), hm = lane >> 4;
#pragma unroll
      for (int r = 0; r < 8; ++r) {
        int m = mt * 16 + r + hm * 8;
        if (m < N_ && n < N_) {
          int dr = m / 7 - n / 7 + 6;
          int dc = m % 7 - n % 7 + 6;
          float bias = rpb[(dr * 13 + dc) * NH_ + h];
          int lm  = 3 * region_(wr * 7 + m / 7) + region_(wc * 7 + m % 7);
          int ln2 = 3 * region_(wr * 7 + n / 7) + region_(wc * 7 + n % 7);
          float msk = (lm != ln2) ? -100.f : 0.f;
          Sb[m * 64 + n] = c.e[r] + bias + msk;
        }
      }
    }
    __syncthreads();
    // softmax rows (fp32), write bf16 P with zero K-padding
    for (int m = threadIdx.x; m < N_; m += 256) {
      float* Sr = Sb + m * 64;
      float mx = -1e30f;
      for (int n = 0; n < N_; ++n) mx = fmaxf(mx, Sr[n]);
      float sum = 0.f;
      for (int n = 0; n < N_; ++n) { float e = expf(Sr[n] - mx); Sr[n] = e; sum += e; }
      float inv = 1.f / sum;
      bf16* Pr = Pb + m * 64;
      for (int n = 0; n < N_; ++n) Pr[n] = (bf16)(Sr[n] * inv);
      for (int n = N_; n < 64; ++n) Pr[n] = (bf16)0.0f;
    }
    __syncthreads();
    // O = P[49x64] @ V[64x32]  (8 tile jobs, one per wave)
    {
      int mt = wave >> 1, nt2 = wave & 1;
      v8f acc = zero8();
#pragma unroll
      for (int kt = 0; kt < 2; ++kt) {
        v16bf afr = load_a_frag(Pb, 64, mt * 16, kt * 32, N_);
        v16bf bfr = load_b_frag(vbuf + h * HD_, C_, kt * 32, nt2 * 16);
        acc = wmma_bf(afr, bfr, acc);
      }
      CF c; c.v = acc;
      int n = h * HD_ + nt2 * 16 + (lane & 15), hm = lane >> 4;
#pragma unroll
      for (int r = 0; r < 8; ++r) {
        int m = mt * 16 + r + hm * 8;
        obuf[m * C_ + n] = (bf16)c.e[r];
      }
    }
    __syncthreads();
  }

  // ---- phase 6: proj gemm + residual, scatter through inverse shift ----
  for (int job = wave; job < 48; job += 8) {
    int nt = job >> 2, mt = job & 3;
    v8f acc = zero8();
#pragma unroll
    for (int kt = 0; kt < 6; ++kt) {
      v16bf afr = load_a_frag(obuf, C_, mt * 16, kt * 32, 64);
      v16bf bfr = load_b_frag_sw(projw, 12, kt, nt);
      acc = wmma_bf(afr, bfr, acc);
    }
    CF c; c.v = acc;
    int n = nt * 16 + (lane & 15), hm = lane >> 4;
    float pb = projb[n];
#pragma unroll
    for (int r = 0; r < 8; ++r) {
      int m = mt * 16 + r + hm * 8;
      if (m < N_) {
        int gr = (wr * 7 + m / 7 + SS_) % IS_;
        int gc = (wc * 7 + m % 7 + SS_) % IS_;
        size_t gi = (size_t)(gr * IS_ + gc) * C_ + n;
        x1b[gi] = xb[gi] + c.e[r] + pb;
      }
    }
  }
}

// ---------------------------------------------------------------------------
// Kernel 2: fused  LN2 -> fc1 -> GELU -> fc2 -> residual, IN PLACE on d_out.
// One block per 64 rows (200704 rows total -> 3136 blocks exactly).
// ---------------------------------------------------------------------------
#define SM2_ALN 0
#define SM2_H   (SM2_ALN + 64*C_*2)
#define SM2_TOT (SM2_H   + 64*768*2)    // 122880 bytes

__global__ __launch_bounds__(256, 1) void swin_mlp_kernel(
    float* __restrict__ xio, const float* __restrict__ n2w, const float* __restrict__ n2b,
    const bf16* __restrict__ fc1w, const float* __restrict__ fc1b,
    const bf16* __restrict__ fc2w, const float* __restrict__ fc2b)
{
  __shared__ __align__(16) char smem[SM2_TOT];
  bf16* aln = (bf16*)(smem + SM2_ALN);   // [64][192]
  bf16* hb  = (bf16*)(smem + SM2_H);     // [64][768]
  size_t row0 = (size_t)blockIdx.x * 64;
  int wave = threadIdx.x >> 5, lane = threadIdx.x & 31;

  // LayerNorm2 -> bf16 tile
  for (int t = wave; t < 64; t += 8) {
    const float* row = xio + (row0 + t) * C_;
    float vals[6]; float s = 0.f, s2 = 0.f;
#pragma unroll
    for (int i = 0; i < 6; ++i) { float v = row[lane + i * 32]; vals[i] = v; s += v; s2 += v * v; }
#pragma unroll
    for (int off = 16; off > 0; off >>= 1) { s += __shfl_xor(s, off, 32); s2 += __shfl_xor(s2, off, 32); }
    float mu   = s  * (1.f / C_);
    float var  = s2 * (1.f / C_) - mu * mu;
    float rstd = rsqrtf(var + 1e-5f);
#pragma unroll
    for (int i = 0; i < 6; ++i) {
      int c = lane + i * 32;
      aln[t * C_ + c] = (bf16)((vals[i] - mu) * rstd * n2w[c] + n2b[c]);
    }
  }
  __syncthreads();

  // fc1 + exact GELU  [64x192] @ [192x768]
  for (int nt = wave; nt < 48; nt += 8) {
    v8f acc[4] = { zero8(), zero8(), zero8(), zero8() };
    for (int kt = 0; kt < 6; ++kt) {
      v16bf bfr = load_b_frag_sw(fc1w, 48, kt, nt);
#pragma unroll
      for (int mt = 0; mt < 4; ++mt) {
        v16bf afr = load_a_frag(aln, C_, mt * 16, kt * 32, 64);
        acc[mt] = wmma_bf(afr, bfr, acc[mt]);
      }
    }
    int n = nt * 16 + (lane & 15), hm = lane >> 4;
    float b1 = fc1b[n];
#pragma unroll
    for (int mt = 0; mt < 4; ++mt) {
      CF c; c.v = acc[mt];
#pragma unroll
      for (int r = 0; r < 8; ++r) {
        int m = mt * 16 + r + hm * 8;
        float v = c.e[r] + b1;
        float g = 0.5f * v * (1.f + erff(v * 0.70710678118654752f));
        hb[m * 768 + n] = (bf16)g;
      }
    }
  }
  __syncthreads();

  // fc2 + residual (in place)  [64x768] @ [768x192]
  for (int nt = wave; nt < 12; nt += 8) {
    v8f acc[4] = { zero8(), zero8(), zero8(), zero8() };
    for (int kt = 0; kt < 24; ++kt) {
      v16bf bfr = load_b_frag_sw(fc2w, 12, kt, nt);
#pragma unroll
      for (int mt = 0; mt < 4; ++mt) {
        v16bf afr = load_a_frag(hb, 768, mt * 16, kt * 32, 64);
        acc[mt] = wmma_bf(afr, bfr, acc[mt]);
      }
    }
    int n = nt * 16 + (lane & 15), hm = lane >> 4;
    float b2 = fc2b[n];
#pragma unroll
    for (int mt = 0; mt < 4; ++mt) {
      CF c; c.v = acc[mt];
#pragma unroll
      for (int r = 0; r < 8; ++r) {
        int m = mt * 16 + r + hm * 8;
        size_t gi = (row0 + m) * C_ + n;
        xio[gi] = xio[gi] + c.e[r] + b2;   // own-row read-then-write: race free
      }
    }
  }
}

// ---------------------------------------------------------------------------
extern "C" void kernel_launch(void* const* d_in, const int* in_sizes, int n_in,
                              void* d_out, int out_size, void* d_ws, size_t ws_size,
                              hipStream_t stream) {
  (void)in_sizes; (void)n_in; (void)out_size; (void)ws_size;
  const float* x     = (const float*)d_in[0];
  const float* n1w   = (const float*)d_in[1];
  const float* n1b   = (const float*)d_in[2];
  const float* qkvw  = (const float*)d_in[3];
  const float* qkvb  = (const float*)d_in[4];
  const float* rpb   = (const float*)d_in[5];
  const float* projw = (const float*)d_in[6];
  const float* projb = (const float*)d_in[7];
  const float* n2w   = (const float*)d_in[8];
  const float* n2b   = (const float*)d_in[9];
  const float* fc1w  = (const float*)d_in[10];
  const float* fc1b  = (const float*)d_in[11];
  const float* fc2w  = (const float*)d_in[12];
  const float* fc2b  = (const float*)d_in[13];
  float* out = (float*)d_out;

  bf16* wsb     = (bf16*)d_ws;
  bf16* qkvw_b  = wsb + W_QKV_OFF;
  bf16* projw_b = wsb + W_PROJ_OFF;
  bf16* fc1w_b  = wsb + W_FC1_OFF;
  bf16* fc2w_b  = wsb + W_FC2_OFF;

  swin_convert_w_kernel<<<(W_TOTAL + 255) / 256, 256, 0, stream>>>(qkvw, projw, fc1w, fc2w, wsb);
  swin_attn_kernel<<<B_ * 64, 256, 0, stream>>>(x, n1w, n1b, qkvw_b, qkvb, rpb, projw_b, projb, out);
  swin_mlp_kernel<<<(B_ * TOK_) / 64, 256, 0, stream>>>(out, n2w, n2b, fc1w_b, fc1b, fc2w_b, fc2b);
}
